// EuclideanRadialBasisFunction_52424370815597
// MI455X (gfx1250) — compile-verified, matrix operand order checked by reading, not run on hardware
//
#include <hip/hip_runtime.h>
#include <hip/hip_bf16.h>
#include <math.h>
#include <stdint.h>

typedef __attribute__((ext_vector_type(16))) __bf16 v16bf;
typedef __attribute__((ext_vector_type(8)))  float  v8f;
typedef __attribute__((ext_vector_type(4)))  unsigned uint32x4;
typedef __attribute__((ext_vector_type(8)))  int      int32x8;

#define BKT 64            // K elements staged per tile
#define LS  72            // LDS row stride in bf16: 64 + 8 pad (TDM pad: 16B per 128B row)
#define BM  128           // block rows (x)
#define BN  256           // block cols (centers)

__device__ __forceinline__ unsigned short f2bf(float f) {
    unsigned u = __builtin_bit_cast(unsigned, f);
    u += 0x7FFFu + ((u >> 16) & 1u);   // round-to-nearest-even
    return (unsigned short)(u >> 16);
}

union FragB16 { uint4 q[2]; v16bf v; };

// Low 32 bits of a generic pointer to __shared__ == LDS byte offset.
#define LDS_OFF(p) ((unsigned)(unsigned long long)(p))

// Issue a 2D TDM tile load: tileH rows x tileW bf16 elements, row stride = rowStride
// elements, destination LDS gets 16B padding after every 128B row (stride 144B).
__device__ __forceinline__ void tdm_load_2d(unsigned ldsOff, const void* gaddr,
                                            unsigned tensorW, unsigned tensorH,
                                            unsigned tileW, unsigned tileH,
                                            unsigned rowStride) {
    unsigned long long ga = (unsigned long long)gaddr;
    uint32x4 g0;
    g0[0] = 1u;                                          // count=1, user mode
    g0[1] = ldsOff;                                      // lds_addr
    g0[2] = (unsigned)(ga & 0xFFFFFFFFu);                // global_addr[31:0]
    g0[3] = (unsigned)((ga >> 32) & 0x1FFFFFFu) | (2u << 30); // addr[56:32] | type=2
    int32x8 g1;
    g1[0] = (int)((1u << 16)            // data_size = 2 bytes
                | (1u << 20)            // pad_enable
                | (4u << 22)            // pad_interval: 32 DWORDs = 128B
                | (3u << 25));          // pad_amount: 4 DWORDs = 16B
    g1[1] = (int)((tensorW & 0xFFFFu) << 16);                         // dim0[15:0]
    g1[2] = (int)(((tensorW >> 16) & 0xFFFFu) | ((tensorH & 0xFFFFu) << 16));
    g1[3] = (int)(((tensorH >> 16) & 0xFFFFu) | (tileW << 16));       // tile_dim0
    g1[4] = (int)(tileH & 0xFFFFu);                                   // tile_dim1
    g1[5] = (int)rowStride;                                           // dim0_stride
    g1[6] = 0;
    g1[7] = 0;
    asm volatile("tensor_load_to_lds %0, %1" :: "s"(g0), "s"(g1) : "memory");
}

// One wave per row: fp32 -> bf16 conversion fused with sum-of-squares.
__global__ __launch_bounds__(256)
void conv_sumsq_kernel(const float* __restrict__ in, unsigned* __restrict__ outb,
                       float* __restrict__ sq, int rows, int cols) {
    int wave = threadIdx.x >> 5;
    int lane = threadIdx.x & 31;
    int row  = blockIdx.x * 8 + wave;
    if (row >= rows) return;
    const float2* p = (const float2*)(in + (size_t)row * cols);
    unsigned*     q = outb + ((size_t)row * cols >> 1);
    const int n2 = cols >> 1;
    float s = 0.f;
    for (int j = lane; j < n2; j += 32) {
        float2 f = p[j];
        s = fmaf(f.x, f.x, fmaf(f.y, f.y, s));
        q[j] = (unsigned)f2bf(f.x) | ((unsigned)f2bf(f.y) << 16);
    }
    #pragma unroll
    for (int off = 16; off; off >>= 1) s += __shfl_xor(s, off, 32);
    if (lane == 0) sq[row] = s;
}

// TDM double-buffered bf16 GEMM + RBF epilogue:
// out[b,o] = max(xsq[b] + csq[o] - 2 * (x_b . c_o), 0)
// Block tile 128x256, 8 waves (2x4), wave tile 64x64 (4x4 WMMA frags).
__global__ __launch_bounds__(256)
void rbf_gemm_tdm(const unsigned short* __restrict__ Xb,
                  const unsigned short* __restrict__ Cb,
                  const float* __restrict__ xsq, const float* __restrict__ csq,
                  float* __restrict__ out, int K, int BROWS, int OUTF) {
    __shared__ __align__(16) unsigned short ldsA[2][BM * LS];
    __shared__ __align__(16) unsigned short ldsB[2][BN * LS];

    const int tid  = threadIdx.x;
    const int wave = tid >> 5;         // 0..7
    const int lane = tid & 31;
    const int wm   = wave >> 2;        // 0..1 : 64-row slab
    const int wn   = wave & 3;         // 0..3 : 64-col slab

    const int blockRow = blockIdx.y * BM;
    const int blockCol = blockIdx.x * BN;

    const int fr    = lane & 15;              // fragment row (A) / col (B)
    const int kHalf = (lane < 16) ? 0 : 8;    // K-chunk select per ISA layout

    v8f acc[4][4] = {};

    const unsigned short* gA = Xb + (size_t)blockRow * K;
    const unsigned short* gB = Cb + (size_t)blockCol * K;
    const int NT = K / BKT;

    if (wave == 0) {
        tdm_load_2d(LDS_OFF(ldsA[0]), gA, K, BROWS, BKT, BM, K);
        tdm_load_2d(LDS_OFF(ldsB[0]), gB, K, OUTF,  BKT, BN, K);
    }

    for (int kt = 0; kt < NT; ++kt) {
        if (wave == 0) __builtin_amdgcn_s_wait_tensorcnt(0);
        __syncthreads();                       // tile kt visible; tile kt-1 readers done
        if (wave == 0 && kt + 1 < NT) {        // prefetch tile kt+1 into other buffer
            const int nb = (kt + 1) & 1;
            tdm_load_2d(LDS_OFF(ldsA[nb]), gA + (size_t)(kt + 1) * BKT, K, BROWS, BKT, BM, K);
            tdm_load_2d(LDS_OFF(ldsB[nb]), gB + (size_t)(kt + 1) * BKT, K, OUTF,  BKT, BN, K);
        }
        const unsigned short* A  = ldsA[kt & 1];
        const unsigned short* Bt = ldsB[kt & 1];

        #pragma unroll
        for (int ks = 0; ks < BKT; ks += 32) {
            const int kk = ks + kHalf;
            FragB16 af[4], bf[4];
            #pragma unroll
            for (int mf = 0; mf < 4; ++mf) {
                const unsigned short* p = &A[(wm * 64 + mf * 16 + fr) * LS + kk];
                af[mf].q[0] = *(const uint4*)(p);
                af[mf].q[1] = *(const uint4*)(p + 16);
            }
            #pragma unroll
            for (int nf = 0; nf < 4; ++nf) {
                const unsigned short* p = &Bt[(wn * 64 + nf * 16 + fr) * LS + kk];
                bf[nf].q[0] = *(const uint4*)(p);
                bf[nf].q[1] = *(const uint4*)(p + 16);
            }
            #pragma unroll
            for (int mf = 0; mf < 4; ++mf)
                #pragma unroll
                for (int nf = 0; nf < 4; ++nf)
                    acc[mf][nf] = __builtin_amdgcn_wmma_f32_16x16x32_bf16(
                        false, af[mf].v, false, bf[nf].v,
                        (short)0, acc[mf][nf], false, false);
        }
    }

    // epilogue: d2 = xsq + csq - 2*cross, clamp at 0
    #pragma unroll
    for (int nf = 0; nf < 4; ++nf) {
        const int col = blockCol + wn * 64 + nf * 16 + fr;
        const float cs = csq[col];
        #pragma unroll
        for (int mf = 0; mf < 4; ++mf) {
            const int rowBase = blockRow + wm * 64 + mf * 16 + kHalf;
            #pragma unroll
            for (int v = 0; v < 8; ++v) {
                const int row = rowBase + v;
                const float d = fmaf(-2.f, acc[mf][nf][v], xsq[row] + cs);
                out[(size_t)row * OUTF + col] = fmaxf(d, 0.f);
            }
        }
    }
}

extern "C" void kernel_launch(void* const* d_in, const int* in_sizes, int n_in,
                              void* d_out, int out_size, void* d_ws, size_t ws_size,
                              hipStream_t stream) {
    const float* x       = (const float*)d_in[0];
    const float* centers = (const float*)d_in[1];
    float* out = (float*)d_out;

    // Derive dims: in_sizes[0]=B*IN, in_sizes[1]=OUT*IN, out_size=B*OUT
    const double a = (double)in_sizes[0], b = (double)in_sizes[1], o = (double)out_size;
    const int IN  = (int)(sqrt(a * b / o) + 0.5);
    const int B   = in_sizes[0] / IN;
    const int OUT = in_sizes[1] / IN;

    // workspace layout: bf16 copies of x and centers, then row norms
    unsigned short* xb = (unsigned short*)d_ws;
    unsigned short* cb = xb + (size_t)B * IN;
    float* xsq = (float*)(cb + (size_t)OUT * IN);
    float* csq = xsq + B;

    conv_sumsq_kernel<<<(B + 7) / 8, 256, 0, stream>>>(x, (unsigned*)xb, xsq, B, IN);
    conv_sumsq_kernel<<<(OUT + 7) / 8, 256, 0, stream>>>(centers, (unsigned*)cb, csq, OUT, IN);

    dim3 grid(OUT / BN, B / BM);
    rbf_gemm_tdm<<<grid, 256, 0, stream>>>(xb, cb, xsq, csq, out, IN, B, OUT);
}
